// RNN_31267361915156
// MI455X (gfx1250) — matrix operand order repaired and checked by reading, not compiled, not run on hardware
//
#include <hip/hip_runtime.h>
#include <math.h>

typedef __attribute__((ext_vector_type(8)))  __bf16 v8bf;
typedef __attribute__((ext_vector_type(16))) __bf16 v16bf;
typedef __attribute__((ext_vector_type(8)))  float  v8f;

#define RNN_VOCAB 128
#define RNN_H     512
#define RNN_OUT   128
#define RNN_B     128
#define RNN_S     512
#define RNN_NWG   16     // persistent workgroups (co-resident, grid-synced)
#define RNN_TPB   512    // 16 waves of 32
#define SROW      520    // LDS row stride in halves (512 data + 8 pad) -> no bank conflicts

// ---------------- prep: bf16-convert + transpose weights, zero h, reset sync ----
__global__ void rnn_prep(const float* __restrict__ Whh,   // [H][H] row-major
                         const float* __restrict__ Why,   // [H][O] row-major
                         __bf16* __restrict__ WhhT,       // [H][H] = Whh^T
                         __bf16* __restrict__ WhyT,       // [O][H] = Why^T
                         __bf16* __restrict__ h0,
                         __bf16* __restrict__ h1,
                         unsigned* __restrict__ cnt) {
  int stride = gridDim.x * blockDim.x;
  int i0 = blockIdx.x * blockDim.x + threadIdx.x;
  for (int i = i0; i < RNN_H * RNN_H; i += stride) {
    int k = i / RNN_H, n = i % RNN_H;
    WhhT[n * RNN_H + k] = (__bf16)Whh[k * RNN_H + n];
  }
  for (int i = i0; i < RNN_H * RNN_OUT; i += stride) {
    int k = i / RNN_OUT, o = i % RNN_OUT;
    WhyT[o * RNN_H + k] = (__bf16)Why[k * RNN_OUT + o];
  }
  for (int i = i0; i < RNN_B * RNN_H; i += stride) {
    h0[i] = (__bf16)0.0f;
    h1[i] = (__bf16)0.0f;
  }
  if (i0 == 0) *cnt = 0u;
}

// ---------------- fragment loaders (ISA 7.12.2 layouts, wave32) ----------------
// A (16x32 bf16, MxK) from LDS-staged rows: lane L holds row M=L&15; lanes<16
// K ∈ {0..7,16..23}, lanes>=16 K ∈ {8..15,24..31}.
__device__ __forceinline__ v16bf load_a_lds(const __bf16* sA, int row, int k0,
                                            int lane) {
  const __bf16* p = sA + row * SROW + k0 + ((lane & 16) ? 8 : 0);
  v8bf lo = *(const v8bf*)(p);
  v8bf hi = *(const v8bf*)(p + 16);
  return __builtin_shufflevector(lo, hi, 0,1,2,3,4,5,6,7,8,9,10,11,12,13,14,15);
}

// B (32x16 bf16, KxN) from PRE-TRANSPOSED weights WT[n][k]: lane L holds column
// N=L&15; lanes<16 K ∈ {0..7,16..23}, lanes>=16 K ∈ {8..15,24..31} (matches A).
__device__ __forceinline__ v16bf load_b_frag(const __bf16* __restrict__ wT,
                                             int col, int k0, int lane) {
  const __bf16* p = wT + col * RNN_H + k0 + ((lane & 16) ? 8 : 0);
  v8bf lo = *(const v8bf*)(p);
  v8bf hi = *(const v8bf*)(p + 16);
  return __builtin_shufflevector(lo, hi, 0,1,2,3,4,5,6,7,8,9,10,11,12,13,14,15);
}

__device__ __forceinline__ v8f wmma_bf16(v16bf a, v16bf b, v8f c) {
  return __builtin_amdgcn_wmma_f32_16x16x32_bf16(false, a, false, b,
                                                 (short)0, c, false, false);
}

// Branch-free tanh: 1 - 2/(e^{2x}+1). Saturates correctly at +/-1 for large |x|
// (exp -> inf or 0), no EXEC manipulation, one v_exp_f32 + one v_rcp_f32.
__device__ __forceinline__ float fast_tanh(float x) {
  float e = __expf(2.0f * x);
  return 1.0f - 2.0f * __builtin_amdgcn_rcpf(e + 1.0f);
}

// ---------------- persistent RNN kernel ---------------------------------------
__global__ void __launch_bounds__(RNN_TPB, 1)
rnn_persistent(const int*   __restrict__ x,     // [B][S] token ids
               const float* __restrict__ Wxh,   // [V][H] fp32 (row gather)
               const float* __restrict__ bh,    // [H]
               const float* __restrict__ by,    // [O]
               const __bf16* __restrict__ WhhT, // [H][H]
               const __bf16* __restrict__ WhyT, // [O][H]
               __bf16* __restrict__ hbuf0,      // [B][H] double buffer
               __bf16* __restrict__ hbuf1,
               float* __restrict__ out,         // [B][S][O]
               unsigned* __restrict__ cnt) {
  __shared__ alignas(16) __bf16 sA[16 * SROW];  // 16 staged H_t rows, padded

  const int tid  = threadIdx.x;
  const int lane = tid & 31;
  const int wave = tid >> 5;
  const int wg   = blockIdx.x;

  // All 16 waves of a WG share one 16-row batch slice of H_t:
  const int mrow0 = (wg >> 1) * 16;               // == htile>>5 == ytile>>3 base
  const int hn0   = ((wg & 1) * 16 + wave) * 16;  // hidden-col base, 0..496

  // y tiles: 64 (8x8), 4 per WG on waves 12..15 (same A rows as h-GEMM)
  const bool has_y = (wave >= 12);                // wave-uniform: EXEC stays full
  const int ytile = wg * 4 + (wave - 12);
  const int yn0 = (ytile & 7) * 16;

  const int l15 = lane & 15;
  const int hiHalf = (lane >> 4) & 1;             // C row = base + r + 8*hiHalf

  // staging map: 512 threads copy 16 rows x 1KB (32B each) global->LDS
  const int srow = tid >> 5;                      // 0..15
  const int scol = tid & 31;                      // 0..31 (2x16B chunks)

  // Iteration t computes H_{t+1} (t<S) and y_{t-1} (t>0); both read only H_t.
  for (int t = 0; t <= RNN_S; ++t) {
    const __bf16* hcur = (t & 1) ? hbuf1 : hbuf0;   // holds H_t
    __bf16*       hnxt = (t & 1) ? hbuf0 : hbuf1;   // receives H_{t+1}

    // ---- stage this WG's 16 A-rows of H_t into LDS (once per WG per step) ----
    {
      const uint4* g = (const uint4*)(hcur + (size_t)(mrow0 + srow) * RNN_H);
      uint4* s = (uint4*)(sA + srow * SROW);
      uint4 a = g[scol * 2];
      uint4 b = g[scol * 2 + 1];
      s[scol * 2]     = a;
      s[scol * 2 + 1] = b;
    }
    __syncthreads();

    if (t < RNN_S) {
      v8f acc;
      #pragma unroll
      for (int i = 0; i < 8; ++i) acc[i] = 0.0f;
      const int bcol = hn0 + l15;
      #pragma unroll 4
      for (int k0 = 0; k0 < RNN_H; k0 += 32) {
        v16bf a = load_a_lds(sA, l15, k0, lane);          // LDS port
        v16bf b = load_b_frag(WhhT, bcol, k0, lane);      // L2 port
        acc = wmma_bf16(a, b, acc);
      }
      // epilogue: + W_xh[token] gather + bias, branch-free tanh, store bf16
      const int n = hn0 + l15;
      const float bias = bh[n];
      #pragma unroll
      for (int r = 0; r < 8; ++r) {
        const int brow = mrow0 + r + hiHalf * 8;
        const int tok = x[brow * RNN_S + t];
        float v = acc[r] + bias + Wxh[tok * RNN_H + n];
        hnxt[brow * RNN_H + n] = (__bf16)fast_tanh(v);
      }
    }

    if (has_y && t > 0) {
      v8f acc;
      #pragma unroll
      for (int i = 0; i < 8; ++i) acc[i] = 0.0f;
      const int bcol = yn0 + l15;
      #pragma unroll 4
      for (int k0 = 0; k0 < RNN_H; k0 += 32) {
        v16bf a = load_a_lds(sA, l15, k0, lane);
        v16bf b = load_b_frag(WhyT, bcol, k0, lane);
        acc = wmma_bf16(a, b, acc);
      }
      const int o = yn0 + l15;
      const float bias = by[o];
      #pragma unroll
      for (int r = 0; r < 8; ++r) {
        const int brow = mrow0 + r + hiHalf * 8;
        out[(size_t)brow * RNN_S * RNN_OUT + (size_t)(t - 1) * RNN_OUT + o] =
            acc[r] + bias;
      }
    }

    // ---- device-wide step barrier: one arrival per WG, per-thread spin -------
    if (t < RNN_S) {
      __threadfence();     // publish this thread's H_{t+1} stores device-wide
      __syncthreads();     // all waves done (also: all sA reads of H_t done)
      if (tid == 0) {
        __hip_atomic_fetch_add(cnt, 1u, __ATOMIC_RELEASE,
                               __HIP_MEMORY_SCOPE_AGENT);
      }
      const unsigned target = (unsigned)(RNN_NWG * (t + 1));
      while (__hip_atomic_load(cnt, __ATOMIC_RELAXED,
                               __HIP_MEMORY_SCOPE_AGENT) < target) {
        __builtin_amdgcn_s_sleep(2);
      }
      __threadfence();     // acquire: drop stale near-cache lines before restage
    }
  }
}

// ---------------- launcher -----------------------------------------------------
extern "C" void kernel_launch(void* const* d_in, const int* in_sizes, int n_in,
                              void* d_out, int out_size, void* d_ws, size_t ws_size,
                              hipStream_t stream) {
  const int*   x   = (const int*)d_in[0];
  const float* Wxh = (const float*)d_in[1];
  const float* Whh = (const float*)d_in[2];
  const float* Why = (const float*)d_in[3];
  const float* bh  = (const float*)d_in[4];
  const float* by  = (const float*)d_in[5];
  float* out = (float*)d_out;

  char* ws = (char*)d_ws;
  const size_t hBytes    = (size_t)RNN_B * RNN_H * sizeof(__bf16);     // 128 KB
  const size_t whhTBytes = (size_t)RNN_H * RNN_H * sizeof(__bf16);     // 512 KB
  const size_t whyTBytes = (size_t)RNN_OUT * RNN_H * sizeof(__bf16);   // 128 KB
  __bf16* h0    = (__bf16*)(ws);
  __bf16* h1    = (__bf16*)(ws + hBytes);
  __bf16* WhhT  = (__bf16*)(ws + 2 * hBytes);
  __bf16* WhyT  = (__bf16*)(ws + 2 * hBytes + whhTBytes);
  unsigned* cnt = (unsigned*)(ws + 2 * hBytes + whhTBytes + whyTBytes);

  rnn_prep<<<512, 256, 0, stream>>>(Whh, Why, WhhT, WhyT, h0, h1, cnt);
  rnn_persistent<<<RNN_NWG, RNN_TPB, 0, stream>>>(x, Wxh, bh, by, WhhT, WhyT,
                                                  h0, h1, out, cnt);
}
